// DHVT_Block_86139864088755
// MI455X (gfx1250) — matrix operand order repaired and checked by reading, not compiled, not run on hardware
//
#include <hip/hip_runtime.h>
#include <hip/hip_bf16.h>

// ---------------------------------------------------------------------------
// DHVT block for MI455X (gfx1250, wave32, WMMA f32_16x16x32_f16,
// async global->LDS staging, pre-swizzled weights)
// ---------------------------------------------------------------------------

typedef __attribute__((ext_vector_type(16))) _Float16 v16h;
typedef __attribute__((ext_vector_type(8)))  _Float16 v8h;
typedef __attribute__((ext_vector_type(8)))  float    v8f;

#define kB   128
#define kN   197
#define kC   384
#define kH   6
#define kHD  64
#define kHID 1536
#define kS   14
#define kM   203
#define kMp  208     // padded token count (13 * 16)
#define kKp  224     // padded key count for P@V (7 * 32)
#define SCALEF 0.125f

__device__ __forceinline__ float dgelu(float x) {
    return 0.5f * x * (1.0f + erff(x * 0.70710678118654752f));
}

__device__ __forceinline__ v8f wmma_f16(v16h a, v16h b, v8f c) {
    return __builtin_amdgcn_wmma_f32_16x16x32_f16(false, a, false, b, (short)0, c, false, false);
}

// ---------------------------------------------------------------------------
// f32 weights [K][N] -> f16 pre-swizzled WMMA B-operand tiles:
// dst[((kt*(N/16) + nt)*32 + lane)*16 + elem]
//   lane = (n&15) | (((k>>4)&1)<<4),  elem = k&15,  kt = k>>5, nt = n>>4
// Each (kt,nt) block is a contiguous 1KB B operand (32 lanes x 16 halves).
// ---------------------------------------------------------------------------
__global__ void cvt_swz_kernel(const float* __restrict__ src, _Float16* __restrict__ dst,
                               int K, int N) {
    int i = blockIdx.x * 256 + threadIdx.x;
    if (i >= K * N) return;
    int k = i / N, n = i % N;
    int kt = k >> 5, nt = n >> 4;
    int lane = (n & 15) | (((k >> 4) & 1) << 4);
    int elem = k & 15;
    dst[(((size_t)kt * (N >> 4) + nt) * 32 + lane) * 16 + elem] = (_Float16)src[i];
}

// mode 0: scale = 1, shift = cb   (plain bias)
// mode 1: scale = g * rsqrt(1+eps), shift = cb*scale + bb   (bias folded into BN)
__global__ void affine_kernel(const float* __restrict__ g, const float* __restrict__ bb,
                              const float* __restrict__ cb, float* __restrict__ s,
                              float* __restrict__ sh, int n, int mode) {
    int i = blockIdx.x * 256 + threadIdx.x;
    if (i >= n) return;
    if (mode == 0) {
        s[i]  = 1.0f;
        sh[i] = cb ? cb[i] : 0.0f;
    } else {
        float ss = g[i] * rsqrtf(1.0f + 1e-5f);
        s[i]  = ss;
        sh[i] = cb[i] * ss + bb[i];
    }
}

// ---------------------------------------------------------------------------
// LN1: x(B,197,384) -> xa_h f16 rows [0..196]; rows 203..207 zeroed (padding).
// one wave per row.
// ---------------------------------------------------------------------------
__global__ __launch_bounds__(256)
void ln1_kernel(const float* __restrict__ x, const float* __restrict__ g,
                const float* __restrict__ bvec, _Float16* __restrict__ xa) {
    int row  = blockIdx.x * 8 + (threadIdx.x >> 5);
    int lane = threadIdx.x & 31;
    int b = row / kMp, n = row % kMp;
    if (b >= kB) return;
    if (n >= kN) {
        if (n >= kM) {  // zero the padded rows
            for (int c = lane; c < kC; c += 32) xa[(size_t)row * kC + c] = (_Float16)0.0f;
        }
        return;  // rows 197..202 written by headtok_kernel
    }
    const float* xr = x + ((size_t)b * kN + n) * kC;
    float v[12];
    float s = 0.0f;
#pragma unroll
    for (int i = 0; i < 12; ++i) { v[i] = xr[lane + i * 32]; s += v[i]; }
#pragma unroll
    for (int off = 16; off; off >>= 1) s += __shfl_xor(s, off, 32);
    float mean = s * (1.0f / kC);
    float s2 = 0.0f;
#pragma unroll
    for (int i = 0; i < 12; ++i) { float d = v[i] - mean; s2 += d * d; }
#pragma unroll
    for (int off = 16; off; off >>= 1) s2 += __shfl_xor(s2, off, 32);
    float inv = rsqrtf(s2 * (1.0f / kC) + 1e-5f);
#pragma unroll
    for (int i = 0; i < 12; ++i) {
        int c = lane + i * 32;
        xa[(size_t)row * kC + c] = (_Float16)((v[i] - mean) * inv * g[c] + bvec[c]);
    }
}

// ---------------------------------------------------------------------------
// head mean: xh0[b][c] = mean over n<197 of xa_h[b][n][c]
// ---------------------------------------------------------------------------
__global__ __launch_bounds__(384)
void headmean_kernel(const _Float16* __restrict__ xa, float* __restrict__ xh0) {
    int b = blockIdx.x, c = threadIdx.x;
    float s = 0.0f;
    for (int n = 0; n < kN; ++n) s += (float)xa[((size_t)b * kMp + n) * kC + c];
    xh0[b * kC + c] = s * (1.0f / kN);
}

// ---------------------------------------------------------------------------
// head token: gelu(LN_64(xh0[b, h*64:..] @ ht_w + ht_b)) + pos -> xa rows 197+h
// ---------------------------------------------------------------------------
__global__ __launch_bounds__(384)
void headtok_kernel(const float* __restrict__ xh0, const float* __restrict__ ht_w,
                    const float* __restrict__ ht_b, const float* __restrict__ htn_g,
                    const float* __restrict__ htn_b, const float* __restrict__ pos,
                    _Float16* __restrict__ xa) {
    int b = blockIdx.x / kH, hh = blockIdx.x % kH;
    int c = threadIdx.x;
    __shared__ float buf[kC];
    const float* xr = xh0 + (size_t)b * kC + hh * kHD;
    float s = ht_b[c];
    for (int d = 0; d < kHD; ++d) s += xr[d] * ht_w[d * kC + c];
    buf[c] = s;
    __syncthreads();
    int g = c >> 6;  // 64-wide chunk
    float m = 0.0f;
    for (int j = 0; j < 64; ++j) m += buf[g * 64 + j];
    m *= (1.0f / 64.0f);
    float var = 0.0f;
    for (int j = 0; j < 64; ++j) { float d = buf[g * 64 + j] - m; var += d * d; }
    var *= (1.0f / 64.0f);
    int d = c & 63;
    float y = (buf[c] - m) * rsqrtf(var + 1e-5f) * htn_g[d] + htn_b[d];
    y = dgelu(y) + pos[hh * kC + c];
    xa[((size_t)b * kMp + kN + hh) * kC + c] = (_Float16)y;
}

// ---------------------------------------------------------------------------
// WMMA GEMM: C[M x N] = act( (A[M x K] @ B[K x N]) * scale[n] + shift[n] )
// A f16 row-major; Bsw f16 pre-swizzled B-operand tiles.
// Block = 256 threads (8 waves), 64x64 C tile, K step 32.
// A tile staged via GLOBAL_LOAD_ASYNC_TO_LDS_B128 (per-lane gather-scatter
// into the WMMA A-operand swizzle), B operands loaded straight from L2.
// ---------------------------------------------------------------------------
template <int ACT, int OUTF16>
__global__ __launch_bounds__(256)
void gemm_wmma_kernel(const _Float16* __restrict__ A, const _Float16* __restrict__ Bsw,
                      const float* __restrict__ scale, const float* __restrict__ shift,
                      float* __restrict__ Cf, _Float16* __restrict__ Ch,
                      int Mrows, int K, int Nc) {
    __shared__ __align__(32) _Float16 lA[4 * 32 * 16];  // [rowtile][lane][16]

    const int t    = threadIdx.x;
    const int lane = t & 31;
    const int wv   = t >> 5;         // 0..7
    const int rt   = wv & 3;         // row tile of this wave
    const int cp   = (wv >> 2) * 2;  // first of two col tiles
    const int m0   = blockIdx.y * 64;
    const int n0   = blockIdx.x * 64;
    const int ntiles = Nc >> 4;

    v8f acc0 = {}, acc1 = {};

    // this thread's fixed 16-byte A chunk: global source row/col and swizzled
    // LDS destination (constant across the K loop).
    const int arow = t >> 2;        // 0..63
    const int akb  = (t & 3) * 8;   // 0,8,16,24
    const int tl = arow >> 4;
    const int ln = (arow & 15) | (((akb >> 4) & 1) << 4);
    const int eb = ((akb >> 3) & 1) * 8;
    const uint32_t ldsdst = (uint32_t)(uintptr_t)(&lA[(tl * 32 + ln) * 16 + eb]);
    const _Float16* agp = A + (size_t)(m0 + arow) * K + akb;

    for (int k0 = 0; k0 < K; k0 += 32) {
        // async gather-scatter copy of the 64x32 A tile into operand layout
        asm volatile("global_load_async_to_lds_b128 %0, %1, off"
                     :: "v"(ldsdst), "v"(agp + k0)
                     : "memory");
        // B operands: contiguous 32B loads from pre-swizzled weights (L2 hit)
        const _Float16* bbase =
            Bsw + (((size_t)(k0 >> 5) * ntiles + (n0 >> 4) + cp) * 32 + lane) * 16;
        v16h b0 = *(const v16h*)(bbase);
        v16h b1 = *(const v16h*)(bbase + 32 * 16);
        if (k0 + 32 < K)
            __builtin_prefetch(agp + k0 + 32, 0, 0);
        asm volatile("s_wait_asynccnt 0x0" ::: "memory");
        __syncthreads();
        v16h a = *(const v16h*)(&lA[(rt * 32 + lane) * 16]);
        acc0 = wmma_f16(a, b0, acc0);
        acc1 = wmma_f16(a, b1, acc1);
        __syncthreads();
    }

    // ---- epilogue ----
    const int l15 = lane & 15, hf = lane >> 4;
    const int nA = n0 + cp * 16 + l15;
    const int nB = nA + 16;
    const float sA = scale[nA], hA = shift[nA];
    const float sB = scale[nB], hB = shift[nB];
#pragma unroll
    for (int r = 0; r < 8; ++r) {
        int m = m0 + rt * 16 + r + hf * 8;
        float y0 = acc0[r] * sA + hA;
        float y1 = acc1[r] * sB + hB;
        if (ACT == 1) { y0 = dgelu(y0); y1 = dgelu(y1); }
        if (OUTF16) {
            Ch[(size_t)m * Nc + nA] = (_Float16)y0;
            Ch[(size_t)m * Nc + nB] = (_Float16)y1;
        } else {
            Cf[(size_t)m * Nc + nA] = y0;
            Cf[(size_t)m * Nc + nB] = y1;
        }
    }
}

// ---------------------------------------------------------------------------
// fused attention: one wave per (b, h, 16-query tile). qkv is f16 [B,208,1152]
// (q at col 0, k at 384, v at 768 within each row). ctx f16 [B,208,384].
// ---------------------------------------------------------------------------
__global__ __launch_bounds__(32)
void attn_kernel(const _Float16* __restrict__ qkv, _Float16* __restrict__ ctx) {
    __shared__ __align__(32) float    sc[16][kKp];
    __shared__ __align__(32) _Float16 pr[16][kKp];
    __shared__ __align__(32) _Float16 vt[kHD][kKp];

    const int lane = threadIdx.x;
    const int qt = blockIdx.x, hh = blockIdx.y, b = blockIdx.z;
    const int hf = lane >> 4, l15 = lane & 15;
    const int q0 = qt * 16;
    const size_t rs = 3 * kC;  // qkv row stride (halves)
    const _Float16* base = qkv + (size_t)b * kMp * rs + hh * kHD;

    // ---- Q A-operands (K dim = 64 -> two 16x32 operands) ----
    const _Float16* qrow = base + (size_t)(q0 + l15) * rs;
    v16h a0, a1;
    {
        v8h x0 = *(const v8h*)(qrow + hf * 16);
        v8h x1 = *(const v8h*)(qrow + 8 + hf * 16);
        v8h x2 = *(const v8h*)(qrow + 32 + hf * 16);
        v8h x3 = *(const v8h*)(qrow + 40 + hf * 16);
#pragma unroll
        for (int i = 0; i < 8; ++i) {
            a0[i] = x0[i]; a0[i + 8] = x1[i];
            a1[i] = x2[i]; a1[i + 8] = x3[i];
        }
    }

    // ---- stage V^T into LDS: vt[d][key] ----
    const _Float16* vbase = base + 2 * kC;
    for (int k = lane; k < kMp; k += 32) {
        const _Float16* vrow = vbase + (size_t)k * rs;
#pragma unroll
        for (int d8 = 0; d8 < 8; ++d8) {
            v8h vv = *(const v8h*)(vrow + d8 * 8);
#pragma unroll
            for (int j = 0; j < 8; ++j) vt[d8 * 8 + j][k] = vv[j];
        }
    }
    for (int d = lane; d < kHD; d += 32)
        for (int k = kMp; k < kKp; ++k) vt[d][k] = (_Float16)0.0f;
    __syncthreads();

    // ---- scores: S = Q @ K^T * scale ----
    const _Float16* kbase = base + kC;
    for (int kt = 0; kt < 13; ++kt) {
        const _Float16* krow = kbase + (size_t)(kt * 16 + l15) * rs;
        v16h b0 = *(const v16h*)(krow + hf * 16);
        v16h b1 = *(const v16h*)(krow + 32 + hf * 16);
        v8f c = {};
        c = wmma_f16(a0, b0, c);
        c = wmma_f16(a1, b1, c);
#pragma unroll
        for (int r = 0; r < 8; ++r) sc[r + hf * 8][kt * 16 + l15] = c[r] * SCALEF;
    }
    __syncthreads();

    // ---- masked softmax (keys >= 203 excluded); row l15, halves split cols ----
    {
        int row = l15;
        int ks = hf ? 102 : 0;
        int ke = hf ? kM : 102;
        float mx = -1e30f;
        for (int k = ks; k < ke; ++k) mx = fmaxf(mx, sc[row][k]);
        mx = fmaxf(mx, __shfl_xor(mx, 16, 32));
        float sum = 0.0f;
        for (int k = ks; k < ke; ++k) sum += __expf(sc[row][k] - mx);
        sum += __shfl_xor(sum, 16, 32);
        float inv = 1.0f / sum;
        for (int k = ks; k < ke; ++k) pr[row][k] = (_Float16)(__expf(sc[row][k] - mx) * inv);
        int zs = kM + hf * 11;
        int ze = hf ? kKp : (kM + 11);
        for (int k = zs; k < ze; ++k) pr[row][k] = (_Float16)0.0f;
    }
    __syncthreads();

    // ---- out = P(16x224) @ V(224x64) ----
    v8f o[4] = {};
    for (int ks = 0; ks < kKp; ks += 32) {
        v16h a;
        v8h p0 = *(const v8h*)(&pr[l15][ks + hf * 16]);
        v8h p1 = *(const v8h*)(&pr[l15][ks + 8 + hf * 16]);
#pragma unroll
        for (int i = 0; i < 8; ++i) { a[i] = p0[i]; a[i + 8] = p1[i]; }
#pragma unroll
        for (int nt = 0; nt < 4; ++nt) {
            v16h bb = *(const v16h*)(&vt[nt * 16 + l15][ks + hf * 16]);
            o[nt] = wmma_f16(a, bb, o[nt]);
        }
    }

    // ---- store ctx ----
#pragma unroll
    for (int nt = 0; nt < 4; ++nt) {
#pragma unroll
        for (int r = 0; r < 8; ++r) {
            int m = q0 + r + hf * 8;
            int n = nt * 16 + l15;
            ctx[((size_t)b * kMp + m) * kC + hh * kHD + n] = (_Float16)o[nt][r];
        }
    }
}

// ---------------------------------------------------------------------------
// residual 1: x1 = x + [cls + mean(head rows); rows 1..196]
// ---------------------------------------------------------------------------
__global__ __launch_bounds__(384)
void resid1_kernel(const float* __restrict__ x, const float* __restrict__ ao,
                   float* __restrict__ x1) {
    int b = blockIdx.x, c = threadIdx.x;
    float hm = 0.0f;
    for (int r = kN; r < kM; ++r) hm += ao[((size_t)b * kMp + r) * kC + c];
    hm *= (1.0f / 6.0f);
    x1[((size_t)b * kN) * kC + c] =
        x[((size_t)b * kN) * kC + c] + ao[((size_t)b * kMp) * kC + c] + hm;
    for (int n = 1; n < kN; ++n)
        x1[((size_t)b * kN + n) * kC + c] =
            x[((size_t)b * kN + n) * kC + c] + ao[((size_t)b * kMp + n) * kC + c];
}

// ---------------------------------------------------------------------------
// LN2: row 0 -> cls2 (f32), rows 1..196 -> token f16 buffer [B*196, 384]
// ---------------------------------------------------------------------------
__global__ __launch_bounds__(256)
void ln2_kernel(const float* __restrict__ x1, const float* __restrict__ g,
                const float* __restrict__ bvec, float* __restrict__ cls2,
                _Float16* __restrict__ tok) {
    int row  = blockIdx.x * 8 + (threadIdx.x >> 5);
    int lane = threadIdx.x & 31;
    if (row >= kB * kN) return;
    int b = row / kN, n = row % kN;
    const float* xr = x1 + (size_t)row * kC;
    float v[12];
    float s = 0.0f;
#pragma unroll
    for (int i = 0; i < 12; ++i) { v[i] = xr[lane + i * 32]; s += v[i]; }
#pragma unroll
    for (int off = 16; off; off >>= 1) s += __shfl_xor(s, off, 32);
    float mean = s * (1.0f / kC);
    float s2 = 0.0f;
#pragma unroll
    for (int i = 0; i < 12; ++i) { float d = v[i] - mean; s2 += d * d; }
#pragma unroll
    for (int off = 16; off; off >>= 1) s2 += __shfl_xor(s2, off, 32);
    float inv = rsqrtf(s2 * (1.0f / kC) + 1e-5f);
#pragma unroll
    for (int i = 0; i < 12; ++i) {
        int c = lane + i * 32;
        float y = (v[i] - mean) * inv * g[c] + bvec[c];
        if (n == 0) cls2[(size_t)b * kC + c] = y;
        else        tok[((size_t)b * 196 + n - 1) * kC + c] = (_Float16)y;
    }
}

// ---------------------------------------------------------------------------
// depthwise 3x3 conv + BN + GELU + residual: h2 = h1 + gelu(dw*s2 + sh2)
// h1/h2 f16 [B*196, 1536] (token-major, SAME padding on 14x14)
// ---------------------------------------------------------------------------
__global__ __launch_bounds__(256)
void dwconv_kernel(const _Float16* __restrict__ h1, const float* __restrict__ w9,
                   const float* __restrict__ s2, const float* __restrict__ sh2,
                   _Float16* __restrict__ h2) {
    int p = blockIdx.x % 196;
    int b = blockIdx.x / 196;
    int yy = p / kS, xx = p % kS;
    for (int ch = threadIdx.x; ch < kHID; ch += 256) {
        float acc = 0.0f;
#pragma unroll
        for (int ky = -1; ky <= 1; ++ky) {
#pragma unroll
            for (int kx = -1; kx <= 1; ++kx) {
                int ny = yy + ky, nx = xx + kx;
                if (ny < 0 || ny >= kS || nx < 0 || nx >= kS) continue;
                float wv = w9[ch * 9 + (ky + 1) * 3 + (kx + 1)];
                acc += wv * (float)h1[((size_t)b * 196 + ny * kS + nx) * kHID + ch];
            }
        }
        float base = (float)h1[((size_t)b * 196 + p) * kHID + ch];
        float z = acc * s2[ch] + sh2[ch];
        h2[((size_t)b * 196 + p) * kHID + ch] = (_Float16)(base + dgelu(z));
    }
}

// ---------------------------------------------------------------------------
// squeeze/excite gate: wbuf[b][c] = gelu(mean(y) @ cmp_w + cmp_b) @ exc_w + exc_b
// ---------------------------------------------------------------------------
__global__ __launch_bounds__(384)
void se_kernel(const float* __restrict__ yb, const float* __restrict__ cmp_w,
               const float* __restrict__ cmp_b, const float* __restrict__ exc_w,
               const float* __restrict__ exc_b, float* __restrict__ wbuf) {
    int b = blockIdx.x, c = threadIdx.x;
    __shared__ float mv[kC];
    __shared__ float z[96];
    float s = 0.0f;
    for (int p = 0; p < 196; ++p) s += yb[((size_t)b * 196 + p) * kC + c];
    mv[c] = s * (1.0f / 196.0f);
    __syncthreads();
    if (c < 96) {
        float a = cmp_b[c];
        for (int j = 0; j < kC; ++j) a += mv[j] * cmp_w[j * 96 + c];
        z[c] = dgelu(a);
    }
    __syncthreads();
    float o = exc_b[c];
    for (int j = 0; j < 96; ++j) o += z[j] * exc_w[j * kC + c];
    wbuf[(size_t)b * kC + c] = o;
}

// ---------------------------------------------------------------------------
// final: out = x1 + concat(cls2*w, tokens)
// ---------------------------------------------------------------------------
__global__ void final_kernel(const float* __restrict__ x1, const float* __restrict__ cls2,
                             const float* __restrict__ wbuf, const float* __restrict__ yb,
                             float* __restrict__ out) {
    size_t i = (size_t)blockIdx.x * 256 + threadIdx.x;
    if (i >= (size_t)kB * kN * kC) return;
    int c = (int)(i % kC);
    size_t row = i / kC;
    int n = (int)(row % kN);
    int b = (int)(row / kN);
    float v;
    if (n == 0)
        v = x1[i] + cls2[(size_t)b * kC + c] * wbuf[(size_t)b * kC + c];
    else
        v = x1[i] + yb[((size_t)b * 196 + n - 1) * kC + c];
    out[i] = v;
}

// ---------------------------------------------------------------------------
// launcher
// ---------------------------------------------------------------------------
extern "C" void kernel_launch(void* const* d_in, const int* in_sizes, int n_in,
                              void* d_out, int out_size, void* d_ws, size_t ws_size,
                              hipStream_t stream) {
    (void)in_sizes; (void)n_in; (void)out_size; (void)ws_size;
    const float* x      = (const float*)d_in[0];
    const float* ln1_g  = (const float*)d_in[1];
    const float* ln1_b  = (const float*)d_in[2];
    const float* qkv_w  = (const float*)d_in[3];
    const float* qkv_b  = (const float*)d_in[4];
    const float* proj_w = (const float*)d_in[5];
    const float* proj_b = (const float*)d_in[6];
    const float* ht_w   = (const float*)d_in[7];
    const float* ht_b   = (const float*)d_in[8];
    const float* htn_g  = (const float*)d_in[9];
    const float* htn_b  = (const float*)d_in[10];
    const float* pos    = (const float*)d_in[11];
    const float* ln2_g  = (const float*)d_in[12];
    const float* ln2_b  = (const float*)d_in[13];
    const float* c1_w   = (const float*)d_in[14];
    const float* c1_b   = (const float*)d_in[15];
    const float* bn1_g  = (const float*)d_in[16];
    const float* bn1_b  = (const float*)d_in[17];
    const float* c2_w   = (const float*)d_in[18];
    const float* c2_b   = (const float*)d_in[19];
    const float* bn2_g  = (const float*)d_in[20];
    const float* bn2_b  = (const float*)d_in[21];
    const float* c3_w   = (const float*)d_in[22];
    const float* c3_b   = (const float*)d_in[23];
    const float* bn3_g  = (const float*)d_in[24];
    const float* bn3_b  = (const float*)d_in[25];
    const float* cmp_w  = (const float*)d_in[26];
    const float* cmp_b  = (const float*)d_in[27];
    const float* exc_w  = (const float*)d_in[28];
    const float* exc_b  = (const float*)d_in[29];
    float* out = (float*)d_out;

    char* wsp = (char*)d_ws;
    auto alloc = [&](size_t bytes) -> char* {
        char* p = wsp;
        wsp += (bytes + 255) & ~(size_t)255;
        return p;
    };

    const size_t rowsA = (size_t)kB * kMp;   // 26624
    const size_t rowsT = (size_t)kB * 196;   // 25088

    _Float16* xa_h    = (_Float16*)alloc(rowsA * kC * 2);
    _Float16* qkvw_s  = (_Float16*)alloc((size_t)kC * 3 * kC * 2);
    _Float16* projw_s = (_Float16*)alloc((size_t)kC * kC * 2);
    _Float16* c1w_s   = (_Float16*)alloc((size_t)kC * kHID * 2);
    _Float16* c3w_s   = (_Float16*)alloc((size_t)kHID * kC * 2);
    _Float16* qkv_h   = (_Float16*)alloc(rowsA * 3 * kC * 2);
    _Float16* ctx_h   = (_Float16*)alloc(rowsA * kC * 2);
    float*    ao      = (float*)   alloc(rowsA * kC * 4);
    float*    x1      = (float*)   alloc((size_t)kB * kN * kC * 4);
    float*    cls2    = (float*)   alloc((size_t)kB * kC * 4);
    _Float16* tok_h   = (_Float16*)alloc(rowsT * kC * 2);
    _Float16* h1_h    = (_Float16*)alloc(rowsT * kHID * 2);
    _Float16* h2_h    = (_Float16*)alloc(rowsT * kHID * 2);
    float*    yb      = (float*)   alloc(rowsT * kC * 4);
    float*    xh0     = (float*)   alloc((size_t)kB * kC * 4);
    float*    wbuf    = (float*)   alloc((size_t)kB * kC * 4);
    float*    sc_qkv  = (float*)alloc(3 * kC * 4);
    float*    sh_qkv  = (float*)alloc(3 * kC * 4);
    float*    sc_proj = (float*)alloc(kC * 4);
    float*    sh_proj = (float*)alloc(kC * 4);
    float*    sc_c1   = (float*)alloc(kHID * 4);
    float*    sh_c1   = (float*)alloc(kHID * 4);
    float*    sc_c3   = (float*)alloc(kC * 4);
    float*    sh_c3   = (float*)alloc(kC * 4);
    float*    s2v     = (float*)alloc(kHID * 4);
    float*    sh2v    = (float*)alloc(kHID * 4);

    // weight conversions: f32 row-major -> f16 pre-swizzled WMMA B tiles
    cvt_swz_kernel<<<(kC * 3 * kC + 255) / 256, 256, 0, stream>>>(qkv_w, qkvw_s, kC, 3 * kC);
    cvt_swz_kernel<<<(kC * kC + 255) / 256, 256, 0, stream>>>(proj_w, projw_s, kC, kC);
    cvt_swz_kernel<<<(kC * kHID + 255) / 256, 256, 0, stream>>>(c1_w, c1w_s, kC, kHID);
    cvt_swz_kernel<<<(kHID * kC + 255) / 256, 256, 0, stream>>>(c3_w, c3w_s, kHID, kC);

    // epilogue affine vectors
    affine_kernel<<<(3 * kC + 255) / 256, 256, 0, stream>>>(nullptr, nullptr, qkv_b, sc_qkv, sh_qkv, 3 * kC, 0);
    affine_kernel<<<(kC + 255) / 256, 256, 0, stream>>>(nullptr, nullptr, proj_b, sc_proj, sh_proj, kC, 0);
    affine_kernel<<<(kHID + 255) / 256, 256, 0, stream>>>(bn1_g, bn1_b, c1_b, sc_c1, sh_c1, kHID, 1);
    affine_kernel<<<(kC + 255) / 256, 256, 0, stream>>>(bn3_g, bn3_b, c3_b, sc_c3, sh_c3, kC, 1);
    affine_kernel<<<(kHID + 255) / 256, 256, 0, stream>>>(bn2_g, bn2_b, c2_b, s2v, sh2v, kHID, 1);

    // LN1 + head-token path -> xa_h (B,208,384) f16
    ln1_kernel<<<(kB * kMp) / 8, 256, 0, stream>>>(x, ln1_g, ln1_b, xa_h);
    headmean_kernel<<<kB, kC, 0, stream>>>(xa_h, xh0);
    headtok_kernel<<<kB * kH, kC, 0, stream>>>(xh0, ht_w, ht_b, htn_g, htn_b, pos, xa_h);

    // QKV GEMM: (26624 x 384) @ (384 x 1152) -> f16
    gemm_wmma_kernel<0, 1><<<dim3(3 * kC / 64, rowsA / 64), 256, 0, stream>>>(
        xa_h, qkvw_s, sc_qkv, sh_qkv, nullptr, qkv_h, (int)rowsA, kC, 3 * kC);

    // attention
    attn_kernel<<<dim3(13, kH, kB), 32, 0, stream>>>(qkv_h, ctx_h);

    // proj GEMM: (26624 x 384) @ (384 x 384) -> f32
    gemm_wmma_kernel<0, 0><<<dim3(kC / 64, rowsA / 64), 256, 0, stream>>>(
        ctx_h, projw_s, sc_proj, sh_proj, ao, nullptr, (int)rowsA, kC, kC);

    // residual + LN2
    resid1_kernel<<<kB, kC, 0, stream>>>(x, ao, x1);
    ln2_kernel<<<(kB * kN) / 8, 256, 0, stream>>>(x1, ln2_g, ln2_b, cls2, tok_h);

    // c1 GEMM with BN+GELU epilogue: (25088 x 384) @ (384 x 1536) -> f16
    gemm_wmma_kernel<1, 1><<<dim3(kHID / 64, rowsT / 64), 256, 0, stream>>>(
        tok_h, c1w_s, sc_c1, sh_c1, nullptr, h1_h, (int)rowsT, kC, kHID);

    // depthwise conv + BN + GELU + residual
    dwconv_kernel<<<kB * 196, 256, 0, stream>>>(h1_h, c2_w, s2v, sh2v, h2_h);

    // c3 GEMM with BN epilogue: (25088 x 1536) @ (1536 x 384) -> f32
    gemm_wmma_kernel<0, 0><<<dim3(kC / 64, rowsT / 64), 256, 0, stream>>>(
        h2_h, c3w_s, sc_c3, sh_c3, yb, nullptr, (int)rowsT, kHID, kC);

    // squeeze-excite gate + final residual assembly
    se_kernel<<<kB, kC, 0, stream>>>(yb, cmp_w, cmp_b, exc_w, exc_b, wbuf);
    final_kernel<<<(kB * kN * kC) / 256, 256, 0, stream>>>(x1, cls2, wbuf, yb, out);
}
